// DLRALayerConv_72851235275280
// MI455X (gfx1250) — matrix operand order repaired and checked by reading, not compiled, not run on hardware
//
#include <hip/hip_runtime.h>
#include <hip/hip_bf16.h>

typedef __attribute__((ext_vector_type(16))) _Float16 v16h;
typedef __attribute__((ext_vector_type(8)))  _Float16 v8h;
typedef __attribute__((ext_vector_type(8)))  float    v8f;

union Frag {
    v16h v;
    v8h  h[2];
};

#define HH 224
#define WW 224
#define CC 64
#define MT 112           // pixels per workgroup tile (224 = 2*112)
#define ZROW 136         // 128 + 8 halves pad to avoid LDS bank conflicts
#define XS_ELEMS (3*114*64)   // 21888 halves = 43776 B

// ---------------- prep kernel 1: W23 = (step==2 ? s @ aux_Vtnp1 : aux_Vt / l_t)
__global__ __launch_bounds__(256) void prep_w23(
    const float* __restrict__ l_t, const float* __restrict__ s,
    const float* __restrict__ aux_Vt, const float* __restrict__ aux_Vtnp1,
    const int* __restrict__ stepp, float* __restrict__ W23)
{
    int idx = blockIdx.x * 256 + threadIdx.x;   // 100*256 total
    int r = idx >> 8, f = idx & 255;
    int st = *stepp;
    float v;
    if (st == 2) {
        float acc = 0.f;
        for (int q = 0; q < 100; ++q)
            acc += s[r * 100 + q] * aux_Vtnp1[q * 256 + f];
        v = acc;
    } else if (st == 1) {
        v = l_t[idx];
    } else {
        v = aux_Vt[idx];
    }
    W23[idx] = v;
}

// ---------------- prep kernel 2: pack U (576x100, step-selected) into B-fragments
__global__ __launch_bounds__(256) void pack_u(
    const float* __restrict__ k, const float* __restrict__ aux_U,
    const float* __restrict__ aux_Unp1, const int* __restrict__ stepp,
    _Float16* __restrict__ Upk)
{
    int idx = blockIdx.x * 256 + threadIdx.x;   // 7*18*512 = 64512 total
    int frag = idx >> 9, w = idx & 511;
    int lane = w >> 4, j = w & 15;
    int nt = frag / 18, ks = frag % 18;
    int n = nt * 16 + (lane & 15);
    int K = ks * 32 + (lane >> 4) * 16 + j;
    int st = *stepp;
    const float* U = (st == 0) ? k : ((st == 1) ? aux_U : aux_Unp1);
    float v = (n < 100) ? U[K * 100 + n] : 0.f;
    Upk[idx] = (_Float16)v;
}

// ---------------- prep kernel 3: pack W23 (100x256, K padded to 128) into B-fragments
__global__ __launch_bounds__(256) void pack_w(
    const float* __restrict__ W23, _Float16* __restrict__ Wpk)
{
    int idx = blockIdx.x * 256 + threadIdx.x;   // 16*4*512 = 32768 total
    int frag = idx >> 9, w = idx & 511;
    int lane = w >> 4, j = w & 15;
    int nt2 = frag >> 2, ks2 = frag & 3;
    int n = nt2 * 16 + (lane & 15);
    int K = ks2 * 32 + (lane >> 4) * 16 + j;
    float v = (K < 100) ? W23[K * 256 + n] : 0.f;
    Wpk[idx] = (_Float16)v;
}

// ---------------- main fused kernel: patches -> GEMM1 -> (LDS) -> GEMM2 -> bias+relu
__global__ __launch_bounds__(224) void dlra_main(
    const float* __restrict__ x, const _Float16* __restrict__ Upk,
    const _Float16* __restrict__ Wpk, const float* __restrict__ bglob,
    const float* __restrict__ aux_bglob, const int* __restrict__ stepp,
    float* __restrict__ out)
{
    __shared__ __align__(16) char smem[XS_ELEMS * 2];   // 43776 B; xs then reused as zs
    _Float16* xs = reinterpret_cast<_Float16*>(smem);
    _Float16* zs = reinterpret_cast<_Float16*>(smem);

    const int tid = threadIdx.x;
    const int g = blockIdx.x;                 // 8*224*2 = 3584 workgroups
    const int b  = g / (HH * 2);
    const int rm = g % (HH * 2);
    const int h  = rm >> 1;
    const int w0 = (rm & 1) * MT;

    const float* bias = (*stepp == 2) ? bglob : aux_bglob;

    const int wave = tid >> 5;
    const int lane = tid & 31;
    const int hi   = lane >> 4;
    const int ml   = lane & 15;
    const int i0   = wave * 16;        // this wave's 16-row M sub-tile
    const int i    = i0 + ml;          // local pixel row for A fragments

    // ---- stage x halo tile (3 rows x 114 cols x 64 ch) as f16 into LDS
    for (int idx = tid; idx < XS_ELEMS; idx += 224) {
        int c   = idx & 63;
        int col = (idx >> 6) % 114;
        int kh  = idx / (114 * 64);
        int gr  = h - 1 + kh;
        int gw  = w0 - 1 + col;
        float v = 0.f;
        if ((unsigned)gr < HH && (unsigned)gw < WW)
            v = x[(((b * HH + gr) * WW + gw) << 6) + c];
        xs[idx] = (_Float16)v;
    }

    // ---- GEMM1 fragment loaders (double-buffered software pipeline)
    Frag a[2];
    Frag bfr[2][7];

    auto loadB1 = [&](Frag* dst, int ks) {
#pragma unroll
        for (int nt = 0; nt < 7; ++nt) {
            const _Float16* bp = Upk + (((nt * 18 + ks) << 9) + lane * 16);
            dst[nt].h[0] = *reinterpret_cast<const v8h*>(bp);
            dst[nt].h[1] = *reinterpret_cast<const v8h*>(bp + 8);
        }
    };
    auto loadA1 = [&](Frag& dst, int ks) {
        int kg = ks * 32;
        int kh = kg / 192;
        int r2 = kg - kh * 192;
        int kw = r2 >> 6;
        int c0 = r2 & 63;
        int abase = ((kh * 114 + (i + kw)) << 6) + c0 + hi * 16;
        dst.h[0] = *reinterpret_cast<const v8h*>(&xs[abase]);
        dst.h[1] = *reinterpret_cast<const v8h*>(&xs[abase + 8]);
    };

    // prefetch first B bank before the barrier: overlaps with x staging + barrier
    loadB1(bfr[0], 0);

    __syncthreads();

    loadA1(a[0], 0);

    // ---- GEMM1: Z(112x112) = patches(112x576) @ U(576x112pad)
    v8f acc[7];
#pragma unroll
    for (int t = 0; t < 7; ++t) acc[t] = v8f{0.f,0.f,0.f,0.f,0.f,0.f,0.f,0.f};

#pragma unroll
    for (int ks = 0; ks < 18; ++ks) {
        const int cur = ks & 1, nxt = cur ^ 1;
        if (ks + 1 < 18) {
            loadB1(bfr[nxt], ks + 1);   // 14-load clause for next K-step
            loadA1(a[nxt], ks + 1);
        }
#pragma unroll
        for (int nt = 0; nt < 7; ++nt) {
            acc[nt] = __builtin_amdgcn_wmma_f32_16x16x32_f16(
                false, a[cur].v, false, bfr[cur][nt].v, (short)0, acc[nt],
                false, false);
        }
    }
    __syncthreads();   // all waves done reading xs; safe to overwrite with zs

    // ---- spill Z to LDS as f16 (A-fragment-friendly layout, padded rows)
#pragma unroll
    for (int nt = 0; nt < 7; ++nt) {
        int kcol = nt * 16 + ml;
#pragma unroll
        for (int r = 0; r < 8; ++r) {
            int row = i0 + r + hi * 8;
            zs[row * ZROW + kcol] = (_Float16)acc[nt][r];
        }
    }
    {   // zero K-pad cols 112..127 of this wave's rows
        v8h z = v8h{0,0,0,0,0,0,0,0};
        *reinterpret_cast<v8h*>(&zs[(i0 + ml) * ZROW + 112 + hi * 8]) = z;
    }
    __syncthreads();

    // ---- GEMM2: out(112x256) = Z(112x128pad) @ W23(128x256), + bias, relu
    Frag b2[2][4];
    auto loadB2 = [&](Frag* dst, int t) {      // t = ch*4 + ks2
        int ch = t >> 2, ks2 = t & 3;
#pragma unroll
        for (int j = 0; j < 4; ++j) {
            int nt2 = ch * 4 + j;
            const _Float16* bp = Wpk + (((nt2 * 4 + ks2) << 9) + lane * 16);
            dst[j].h[0] = *reinterpret_cast<const v8h*>(bp);
            dst[j].h[1] = *reinterpret_cast<const v8h*>(bp + 8);
        }
    };
    auto loadA2 = [&](Frag& dst, int ks2) {
        int abase = i * ZROW + ks2 * 32 + hi * 16;
        dst.h[0] = *reinterpret_cast<const v8h*>(&zs[abase]);
        dst.h[1] = *reinterpret_cast<const v8h*>(&zs[abase + 8]);
    };

    loadB2(b2[0], 0);
    loadA2(a[0], 0);

    const int prow = (b * HH + h) * WW + w0;
#pragma unroll
    for (int ch = 0; ch < 4; ++ch) {
        v8f oc[4];
#pragma unroll
        for (int j = 0; j < 4; ++j) oc[j] = v8f{0.f,0.f,0.f,0.f,0.f,0.f,0.f,0.f};

#pragma unroll
        for (int ks2 = 0; ks2 < 4; ++ks2) {
            const int t = ch * 4 + ks2;
            const int cur = t & 1, nxt = cur ^ 1;
            if (t + 1 < 16) {
                loadB2(b2[nxt], t + 1);     // 8-load clause for next step
                loadA2(a[nxt], (t + 1) & 3);
            }
#pragma unroll
            for (int j = 0; j < 4; ++j) {
                oc[j] = __builtin_amdgcn_wmma_f32_16x16x32_f16(
                    false, a[cur].v, false, b2[cur][j].v, (short)0, oc[j],
                    false, false);
            }
        }
#pragma unroll
        for (int j = 0; j < 4; ++j) {
            int f = (ch * 4 + j) * 16 + ml;
#pragma unroll
            for (int r = 0; r < 8; ++r) {
                int lrow = i0 + r + hi * 8;
                int p    = prow + lrow;
                float v  = oc[j][r] + bias[((h * WW + (w0 + lrow)) << 8) + f];
                out[(p << 8) + f] = v > 0.f ? v : 0.f;
            }
        }
    }
}

extern "C" void kernel_launch(void* const* d_in, const int* in_sizes, int n_in,
                              void* d_out, int out_size, void* d_ws, size_t ws_size,
                              hipStream_t stream) {
    const float* x         = (const float*)d_in[0];
    const float* k         = (const float*)d_in[1];
    const float* l_t       = (const float*)d_in[2];
    const float* s         = (const float*)d_in[3];
    const float* aux_U     = (const float*)d_in[4];
    const float* aux_Unp1  = (const float*)d_in[5];
    const float* aux_Vt    = (const float*)d_in[6];
    const float* aux_Vtnp1 = (const float*)d_in[7];
    const float* b         = (const float*)d_in[8];
    const float* aux_b     = (const float*)d_in[9];
    const int*   step      = (const int*)d_in[10];

    char* ws = (char*)d_ws;
    float*    W23 = (float*)ws;                             // 100*256*4  = 102400 B
    _Float16* Upk = (_Float16*)(ws + 102400);               // 7*18*512*2 = 129024 B
    _Float16* Wpk = (_Float16*)(ws + 102400 + 129024);      // 16*4*512*2 =  65536 B

    prep_w23<<<100, 256, 0, stream>>>(l_t, s, aux_Vt, aux_Vtnp1, step, W23);
    pack_u  <<<252, 256, 0, stream>>>(k, aux_U, aux_Unp1, step, Upk);
    pack_w  <<<128, 256, 0, stream>>>(W23, Wpk);

    dlra_main<<<8 * HH * 2, 224, 0, stream>>>(x, Upk, Wpk, b, aux_b, step,
                                              (float*)d_out);
}
